// MyRNN_2_2353642078659
// MI455X (gfx1250) — matrix operand order repaired and checked by reading, not compiled, not run on hardware
//
#include <hip/hip_runtime.h>
#include <hip/hip_bf16.h>
#include <math.h>

typedef __attribute__((ext_vector_type(16))) _Float16 v16h;
typedef __attribute__((ext_vector_type(8)))  _Float16 v8h;
typedef __attribute__((ext_vector_type(8)))  float    v8f;

#define BATCH_ 4096
#define SEQ_   80
#define EMB_   100
#define EPAD_  128
#define H_     64
#define WAVES_PER_BLK 2
#define THREADS_ (WAVES_PER_BLK * 32)

// Hardware tanh on 8 values in one asm block. Consecutive v_tanh_f32 on distinct
// registers are mutually independent, satisfying the TRANS hazard rule (a trans op must be followed by one
// independent op or a nop); only the last one needs the v_nop.
__device__ __forceinline__ v8f tanh8(const v8f& x) {
    float y0, y1, y2, y3, y4, y5, y6, y7;
    asm volatile(
        "v_tanh_f32 %0, %8\n\t"
        "v_tanh_f32 %1, %9\n\t"
        "v_tanh_f32 %2, %10\n\t"
        "v_tanh_f32 %3, %11\n\t"
        "v_tanh_f32 %4, %12\n\t"
        "v_tanh_f32 %5, %13\n\t"
        "v_tanh_f32 %6, %14\n\t"
        "v_tanh_f32 %7, %15\n\t"
        "v_nop"
        : "=&v"(y0), "=&v"(y1), "=&v"(y2), "=&v"(y3),
          "=&v"(y4), "=&v"(y5), "=&v"(y6), "=&v"(y7)
        : "v"(x[0]), "v"(x[1]), "v"(x[2]), "v"(x[3]),
          "v"(x[4]), "v"(x[5]), "v"(x[6]), "v"(x[7]));
    v8f y;
    y[0] = y0; y[1] = y1; y[2] = y2; y[3] = y3;
    y[4] = y4; y[5] = y5; y[6] = y6; y[7] = y7;
    return y;
}

// ---------- B-fragment staging (row-major f32 [Ksrc][64] -> WMMA B layout f16) ----------
// B 32x16 f16 layout (ISA 7.12.2): lane L holds column N = L%16; half i<8 holds K = 8*(L/16)+i, half i>=8 holds K = 16+8*(L/16)+(i-8).
// Fragment (kt,nt) stored as 512 contiguous halves: [lane][i].
__device__ __forceinline__ void stage_B(const float* W, int Ksrc, int ktiles,
                                        _Float16* dst, int tid) {
    const int total = ktiles * 4 * 512;
    for (int e = tid; e < total; e += THREADS_) {
        int frag = e >> 9;
        int within = e & 511;
        int lane = within >> 4;
        int i = within & 15;
        int kt = frag >> 2, nt = frag & 3;
        int K = kt * 32 + ((i >> 3) << 4) + ((lane >> 4) << 3) + (i & 7);
        int N = nt * 16 + (lane & 15);
        float v = (K < Ksrc) ? W[K * H_ + N] : 0.0f;
        dst[e] = (_Float16)v;
    }
}

__device__ __forceinline__ v16h load_B(const _Float16* frags, int kt, int nt, int lane) {
    return *(const v16h*)(frags + (((kt << 2) + nt) << 9) + (lane << 4));
}

// ---------- A fragment load from row-major f16 stage [16][stride] ----------
// A 16x32 f16 layout: lane L holds row M = L%16; halves 0..7 -> K = 8*(L/16)+i,
// halves 8..15 -> K = 16 + 8*(L/16) + (i-8).
__device__ __forceinline__ v16h load_A(const _Float16* stg, int stride, int kbase, int lane) {
    int M = lane & 15;
    int g = (lane >> 4) << 3;
    const _Float16* p = stg + M * stride + kbase + g;
    v8h lo = *(const v8h*)p;
    v8h hi = *(const v8h*)(p + 16);
    v16h r;
#pragma unroll
    for (int i = 0; i < 8; ++i) { r[i] = lo[i]; r[i + 8] = hi[i]; }
    return r;
}

// ---------- C/D (f32 16x16) -> f16 row-major stage [16][64] ----------
// C layout: lane L -> N = L%16; VGPR r -> M = r + 8*(L/16).
__device__ __forceinline__ void store_C16(_Float16* stg, const v8f& c, int nt, int lane) {
    int Mb = (lane >> 4) << 3;
    int N = (lane & 15) + nt * 16;
#pragma unroll
    for (int r = 0; r < 8; ++r) stg[(Mb + r) * H_ + N] = (_Float16)c[r];
}

__device__ __forceinline__ void store_C32(float* stg, const v8f& c, int nt, int lane) {
    int Mb = (lane >> 4) << 3;
    int N = (lane & 15) + nt * 16;
#pragma unroll
    for (int r = 0; r < 8; ++r) stg[(Mb + r) * H_ + N] = c[r];
}

#define LDS_FENCE() asm volatile("s_wait_dscnt 0x0" ::: "memory")

__global__ __launch_bounds__(THREADS_) void rnn_fused_kernel(
    const int* __restrict__ x, const float* __restrict__ emb,
    const float* __restrict__ W1, const float* __restrict__ U1, const float* __restrict__ b1,
    const float* __restrict__ W2, const float* __restrict__ U2, const float* __restrict__ b2,
    const float* __restrict__ Wo, const float* __restrict__ bo,
    float* __restrict__ out)
{
    // Weight fragments (shared by both waves of the block)
    __shared__ __align__(32) _Float16 sW1[4 * 4 * 512]; // K=128(pad from 100), N=64 : 16 KB
    __shared__ __align__(32) _Float16 sU1[2 * 4 * 512]; // 8 KB
    __shared__ __align__(32) _Float16 sW2[2 * 4 * 512]; // 8 KB
    __shared__ __align__(32) _Float16 sU2[2 * 4 * 512]; // 8 KB
    // Per-wave stages: e-stage 16x128 f16 (aliased later as 16x64 f32 for the head)
    __shared__ __align__(32) float sEf[WAVES_PER_BLK][16 * H_];      // 4 KB/wave
    __shared__ __align__(32) _Float16 sH1[WAVES_PER_BLK][16 * H_];   // 2 KB/wave
    __shared__ __align__(32) _Float16 sH2[WAVES_PER_BLK][16 * H_];   // 2 KB/wave

    const int tid = threadIdx.x;
    const int wave = tid >> 5;
    const int lane = tid & 31;
    const int rowBase = (blockIdx.x * WAVES_PER_BLK + wave) * 16;

    _Float16* eStage = (_Float16*)sEf[wave];   // 16 x 128 f16 view
    float*    fStage = sEf[wave];              // 16 x 64 f32 view (post-loop)
    _Float16* h1S = sH1[wave];
    _Float16* h2S = sH2[wave];

    // ---- one-time staging ----
    stage_B(W1, EMB_, 4, sW1, tid);
    stage_B(U1, H_, 2, sU1, tid);
    stage_B(W2, H_, 2, sW2, tid);
    stage_B(U2, H_, 2, sU2, tid);

    // zero h states + e-stage pad columns (100..127)
    for (int e = lane; e < 16 * H_; e += 32) { h1S[e] = (_Float16)0.0f; h2S[e] = (_Float16)0.0f; }
    for (int e = lane; e < 16 * (EPAD_ - EMB_); e += 32) {
        int r = e / (EPAD_ - EMB_), c = EMB_ + e % (EPAD_ - EMB_);
        eStage[r * EPAD_ + c] = (_Float16)0.0f;
    }
    __syncthreads();

    // per-lane column biases (C layout: N = lane%16 + 16*nt)
    float bc1[4], bc2[4];
#pragma unroll
    for (int n = 0; n < 4; ++n) {
        bc1[n] = b1[n * 16 + (lane & 15)];
        bc2[n] = b2[n * 16 + (lane & 15)];
    }

    for (int t = 0; t < SEQ_; ++t) {
        // ---- gather embeddings for 16 rows into e-stage (f16, padded) ----
        {
            int r = lane >> 1;
            int half = lane & 1;
            int idx = x[(rowBase + r) * SEQ_ + t];
            const float* src = emb + (long)idx * EMB_ + half * 50;
            _Float16* dst = eStage + r * EPAD_ + half * 50;
#pragma unroll
            for (int j = 0; j < 25; ++j) {
                float2 v = *(const float2*)(src + 2 * j);
                dst[2 * j] = (_Float16)v.x;
                dst[2 * j + 1] = (_Float16)v.y;
            }
        }
        LDS_FENCE();

        // ---- load A fragments: e_t (4 k-tiles) and h1_{t-1} (2 k-tiles) ----
        v16h eA[4], h1A[2];
#pragma unroll
        for (int kt = 0; kt < 4; ++kt) eA[kt] = load_A(eStage, EPAD_, kt * 32, lane);
#pragma unroll
        for (int kt = 0; kt < 2; ++kt) h1A[kt] = load_A(h1S, H_, kt * 32, lane);

        // ---- layer 1: h1 = tanh(e@W1 + h1@U1 + b1) ----
#pragma unroll
        for (int nt = 0; nt < 4; ++nt) {
            v8f c = {};
#pragma unroll
            for (int kt = 0; kt < 4; ++kt)
                c = __builtin_amdgcn_wmma_f32_16x16x32_f16(false, eA[kt], false,
                        load_B(sW1, kt, nt, lane), (short)0, c, false, false);
#pragma unroll
            for (int kt = 0; kt < 2; ++kt)
                c = __builtin_amdgcn_wmma_f32_16x16x32_f16(false, h1A[kt], false,
                        load_B(sU1, kt, nt, lane), (short)0, c, false, false);
#pragma unroll
            for (int r = 0; r < 8; ++r) c[r] += bc1[nt];
            c = tanh8(c);
            store_C16(h1S, c, nt, lane);
        }
        LDS_FENCE();

        // ---- layer 2: h2 = tanh(h1_t@W2 + h2_{t-1}@U2 + b2) ----
        v16h h1B[2], h2A[2];
#pragma unroll
        for (int kt = 0; kt < 2; ++kt) {
            h1B[kt] = load_A(h1S, H_, kt * 32, lane);
            h2A[kt] = load_A(h2S, H_, kt * 32, lane);
        }
#pragma unroll
        for (int nt = 0; nt < 4; ++nt) {
            v8f c = {};
#pragma unroll
            for (int kt = 0; kt < 2; ++kt)
                c = __builtin_amdgcn_wmma_f32_16x16x32_f16(false, h1B[kt], false,
                        load_B(sW2, kt, nt, lane), (short)0, c, false, false);
#pragma unroll
            for (int kt = 0; kt < 2; ++kt)
                c = __builtin_amdgcn_wmma_f32_16x16x32_f16(false, h2A[kt], false,
                        load_B(sU2, kt, nt, lane), (short)0, c, false, false);
#pragma unroll
            for (int r = 0; r < 8; ++r) c[r] += bc2[nt];
            c = tanh8(c);
            store_C16(h2S, c, nt, lane);
            if (t == SEQ_ - 1) store_C32(fStage, c, nt, lane); // keep f32 for the head
        }
        LDS_FENCE();
    }

    // ---- head: out[m] = sigmoid(h2[m,:] @ Wo + bo) ----
    if (lane < 16) {
        const int M = lane;
        float s = bo[0];
#pragma unroll 8
        for (int j = 0; j < H_; ++j) s += fStage[M * H_ + j] * Wo[j];
        out[rowBase + M] = 1.0f / (1.0f + __expf(-s));
    }
}

extern "C" void kernel_launch(void* const* d_in, const int* in_sizes, int n_in,
                              void* d_out, int out_size, void* d_ws, size_t ws_size,
                              hipStream_t stream) {
    (void)in_sizes; (void)n_in; (void)d_ws; (void)ws_size; (void)out_size;
    const int*   x   = (const int*)d_in[0];
    const float* emb = (const float*)d_in[1];
    const float* W1  = (const float*)d_in[2];
    const float* U1  = (const float*)d_in[3];
    const float* b1  = (const float*)d_in[4];
    const float* W2  = (const float*)d_in[5];
    const float* U2  = (const float*)d_in[6];
    const float* b2  = (const float*)d_in[7];
    const float* Wo  = (const float*)d_in[8];
    const float* bo  = (const float*)d_in[9];
    float* out = (float*)d_out;

    dim3 grid(BATCH_ / (16 * WAVES_PER_BLK));   // 128 blocks
    dim3 block(THREADS_);                        // 2 waves
    rnn_fused_kernel<<<grid, block, 0, stream>>>(x, emb, W1, U1, b1, W2, U2, b2, Wo, bo, out);
}
